// FusedTransformerBlock_91328184582585
// MI455X (gfx1250) — matrix operand order repaired and checked by reading, not compiled, not run on hardware
//
#include <hip/hip_runtime.h>
#include <math.h>

// ---------------------------------------------------------------------------
// Fused transformer block for MI455X (gfx1250), wave32.
// GEMMs + attention on v_wmma_f32_16x16x32_bf16; A/B panels and K/V tiles
// staged in LDS with global_load_async_to_lds_b128 (ASYNCcnt, double buffer).
// Softmax / LayerNorm / residuals in fp32.
// ---------------------------------------------------------------------------

typedef __bf16 bf16;
typedef __attribute__((ext_vector_type(16))) __bf16 v16bf;
typedef __attribute__((ext_vector_type(8)))  __bf16 v8bf;
typedef __attribute__((ext_vector_type(8)))  float  v8f;

#define CB 4
#define CT 4096
#define CD 1024
#define CH 16
#define CDH 64
#define CF 4096
#define CM (CB*CT)          // 16384 rows

union AFrag { v16bf v; v8bf h[2]; };

static __device__ __forceinline__ v8f wmma_bf16(v16bf a, v16bf b, v8f c) {
  return __builtin_amdgcn_wmma_f32_16x16x32_bf16(false, a, false, b, (short)0, c,
                                                 false, false);
}

// Async copy 16 bytes/lane from global to LDS (CDNA5 async path, ASYNCcnt).
static __device__ __forceinline__ void async_b128(void* lds, const void* g) {
  unsigned ldsoff = (unsigned)(unsigned long long)lds;   // low 32b = LDS offset
  asm volatile("global_load_async_to_lds_b128 %0, %1, off"
               :: "v"(ldsoff), "v"((unsigned long long)g)
               : "memory");
}
#define WAIT_ASYNC(n) asm volatile("s_wait_asynccnt " #n ::: "memory")

// ---------------------------------------------------------------------------
// fp32 -> bf16 elementwise convert (x)
// ---------------------------------------------------------------------------
__global__ __launch_bounds__(256) void cvt_kernel(const float* __restrict__ in,
                                                  bf16* __restrict__ out, int n4) {
  int i = blockIdx.x * 256 + threadIdx.x;
  if (i < n4) {
    float4 v = ((const float4*)in)[i];
    out[i * 4 + 0] = (bf16)v.x;
    out[i * 4 + 1] = (bf16)v.y;
    out[i * 4 + 2] = (bf16)v.z;
    out[i * 4 + 3] = (bf16)v.w;
  }
}

// ---------------------------------------------------------------------------
// W [K][N] fp32 -> Wt [N][K] bf16, tiled through LDS
// ---------------------------------------------------------------------------
__global__ __launch_bounds__(256) void transpose_kernel(const float* __restrict__ W,
                                                        bf16* __restrict__ Wt,
                                                        int K, int N) {
  __shared__ __bf16 tile[32][33];
  int nx = blockIdx.x * 32, kx = blockIdx.y * 32;
  int tx = threadIdx.x & 31, ty = threadIdx.x >> 5;   // 32 x 8
#pragma unroll
  for (int j = 0; j < 32; j += 8)
    tile[ty + j][tx] = (bf16)W[(size_t)(kx + ty + j) * N + nx + tx];
  __syncthreads();
#pragma unroll
  for (int j = 0; j < 32; j += 8)
    Wt[(size_t)(nx + ty + j) * K + kx + tx] = tile[tx][ty + j];
}

// ---------------------------------------------------------------------------
// GEMM: C[M][N] = A[M][K](bf16,row-major) * Bt[N][K](bf16,row-major) + bias
// Block tile 64x128 (8 waves, wave tile 16x64), k-step 32. A/B panels staged
// in LDS via async copies, double buffered.
// EPI 0: QKV scatter -> Q[BH][T][DH], K[BH][T][DH], Vt[BH][DH][T]  (bf16)
// EPI 1: out fp32 = acc + bias + res
// EPI 2: out bf16 = gelu_exact(acc + bias)
// ---------------------------------------------------------------------------
template <int EPI>
__global__ __launch_bounds__(256) void gemm_kernel(
    const bf16* __restrict__ A, const bf16* __restrict__ Bt,
    const float* __restrict__ bias, void* __restrict__ out0,
    const float* __restrict__ res, bf16* __restrict__ outK,
    bf16* __restrict__ outV, int M, int N, int K) {
  __shared__ __align__(16) __bf16 As[2][64 * 32];    // 4 KB per buffer
  __shared__ __align__(16) __bf16 Bs[2][128 * 32];   // 8 KB per buffer

  const int tid = threadIdx.x;
  const int lane = tid & 31;
  const int wave = tid >> 5;
  const int wm = wave & 3, wn = wave >> 2;
  const int mblk = blockIdx.y * 64;
  const int nblk = blockIdx.x * 128;
  const int nl = lane & 15;
  const int kh = lane >> 4;

  // staging assignments (per thread: 1x A b128 + 2x B b128 per stage)
  const int ar = tid >> 2, aq = (tid & 3) * 8;        // A: row, 8-elt chunk
  const int br = tid >> 1, bq = (tid & 1) * 16;       // B: row, 16-elt half

  v8f acc[4] = {};

  // prologue: stage 0
  async_b128(&As[0][ar * 32 + aq], A + (size_t)(mblk + ar) * K + aq);
  async_b128(&Bs[0][br * 32 + bq], Bt + (size_t)(nblk + br) * K + bq);
  async_b128(&Bs[0][br * 32 + bq + 8], Bt + (size_t)(nblk + br) * K + bq + 8);

  for (int kk = 0; kk < K; kk += 32) {
    const int cur = (kk >> 5) & 1;
    if (kk + 32 < K) {
      const int kn = kk + 32;
      async_b128(&As[cur ^ 1][ar * 32 + aq], A + (size_t)(mblk + ar) * K + kn + aq);
      async_b128(&Bs[cur ^ 1][br * 32 + bq], Bt + (size_t)(nblk + br) * K + kn + bq);
      async_b128(&Bs[cur ^ 1][br * 32 + bq + 8],
                 Bt + (size_t)(nblk + br) * K + kn + bq + 8);
      WAIT_ASYNC(3);          // in-order: current stage has landed
    } else {
      WAIT_ASYNC(0);
    }
    __syncthreads();

    AFrag a;
    a.h[0] = *(const v8bf*)&As[cur][(wm * 16 + nl) * 32 + kh * 16];
    a.h[1] = *(const v8bf*)&As[cur][(wm * 16 + nl) * 32 + kh * 16 + 8];
#pragma unroll
    for (int s = 0; s < 4; ++s) {
      v16bf b = *(const v16bf*)&Bs[cur][(wn * 64 + s * 16 + nl) * 32 + kh * 16];
      acc[s] = wmma_bf16(a.v, b, acc[s]);
    }
    __syncthreads();          // all reads done before next overwrite
  }

  const int m0 = mblk + wm * 16;
  const int n0 = nblk + wn * 64;
#pragma unroll
  for (int s = 0; s < 4; ++s) {
#pragma unroll
    for (int i = 0; i < 8; ++i) {
      int gr = m0 + i + kh * 8;
      int gc = n0 + s * 16 + nl;
      float v = acc[s][i] + bias[gc];
      if constexpr (EPI == 0) {
        int part = gc >> 10;            // 0=q 1=k 2=v
        int dcol = gc & 1023;
        int hh = dcol >> 6;
        int dh = dcol & 63;
        int b = gr >> 12;               // / T
        int t = gr & 4095;
        bf16 bv = (bf16)v;
        size_t bh = (size_t)(b * CH + hh);
        if (part == 0)
          ((bf16*)out0)[(bh * CT + t) * CDH + dh] = bv;
        else if (part == 1)
          outK[(bh * CT + t) * CDH + dh] = bv;
        else
          outV[(bh * CDH + dh) * CT + t] = bv;      // V stored transposed
      } else if constexpr (EPI == 1) {
        ((float*)out0)[(size_t)gr * N + gc] = v + res[(size_t)gr * N + gc];
      } else {  // EPI == 2 : exact GELU -> bf16
        float g = 0.5f * v * (1.0f + erff(v * 0.70710678118f));
        ((bf16*)out0)[(size_t)gr * N + gc] = (bf16)g;
      }
    }
  }
}

// ---------------------------------------------------------------------------
// Flash attention: block = 8 waves sharing one (b,h); wave owns 16 query rows.
// Streams 32 keys/iter; K tile [32][64] and Vt tile [64][32] staged in LDS
// via async copies (shared by all 8 waves), double buffered.
// Q,K: [BH][T][DH] bf16; Vt: [BH][DH][T] bf16; ctx out: [B*T][D] bf16.
// ---------------------------------------------------------------------------
__global__ __launch_bounds__(256) void attn_kernel(const bf16* __restrict__ Q,
                                                   const bf16* __restrict__ Km,
                                                   const bf16* __restrict__ Vt,
                                                   bf16* __restrict__ ctx) {
  __shared__ __align__(16) __bf16 Ks[2][32 * 64];      // 4 KB per buffer
  __shared__ __align__(16) __bf16 Vs[2][64 * 32];      // 4 KB per buffer
  __shared__ __align__(16) __bf16 lds_p[8][16 * 32];   // per-wave P tile

  const int tid = threadIdx.x;
  const int lane = tid & 31;
  const int wave = tid >> 5;
  const int nl = lane & 15, kh = lane >> 4;

  const int bh = blockIdx.x >> 5;                       // 32 blocks per (b,h)
  const int t0 = (blockIdx.x & 31) * 128 + wave * 16;
  const bf16* Qb = Q + (size_t)bh * CT * CDH;
  const bf16* Kb = Km + (size_t)bh * CT * CDH;
  const bf16* Vb = Vt + (size_t)bh * CDH * CT;
  __bf16* pbuf = &lds_p[wave][0];

  // staging assignments (per thread: 1x K b128 + 1x V b128 per stage)
  const int krow = tid >> 3, kq = (tid & 7) * 8;        // K: 32 rows x 64 elts
  const int vrow = tid >> 2, vq = (tid & 3) * 8;        // V: 64 rows x 32 elts

  AFrag aq[2];
  {
    const bf16* qp = Qb + (size_t)(t0 + nl) * CDH;
    aq[0].h[0] = *(const v8bf*)(qp + kh * 16);
    aq[0].h[1] = *(const v8bf*)(qp + kh * 16 + 8);
    aq[1].h[0] = *(const v8bf*)(qp + 32 + kh * 16);
    aq[1].h[1] = *(const v8bf*)(qp + 32 + kh * 16 + 8);
  }

  v8f accO[4] = {};
  float mrow[8], lrow[8];
#pragma unroll
  for (int i = 0; i < 8; ++i) { mrow[i] = -INFINITY; lrow[i] = 0.0f; }
  const float scale = 0.125f;  // 1/sqrt(64)

  // prologue: stage keys/values [0,32)
  async_b128(&Ks[0][krow * 64 + kq], Kb + (size_t)krow * CDH + kq);
  async_b128(&Vs[0][vrow * 32 + vq], Vb + (size_t)vrow * CT + vq);

  for (int kt = 0; kt < CT; kt += 32) {
    const int cur = (kt >> 5) & 1;
    if (kt + 32 < CT) {
      const int kn = kt + 32;
      async_b128(&Ks[cur ^ 1][krow * 64 + kq], Kb + (size_t)(kn + krow) * CDH + kq);
      async_b128(&Vs[cur ^ 1][vrow * 32 + vq], Vb + (size_t)vrow * CT + kn + vq);
      WAIT_ASYNC(2);
    } else {
      WAIT_ASYNC(0);
    }
    __syncthreads();

    // ---- scores: S[16][32], two 16x16 wmma chains over DH=64 ----
    v8f s[2];
#pragma unroll
    for (int j = 0; j < 2; ++j) {
      const __bf16* kp = &Ks[cur][(j * 16 + nl) * 64];
      v16bf b0 = *(const v16bf*)(kp + kh * 16);
      v16bf b1 = *(const v16bf*)(kp + 32 + kh * 16);
      v8f z = {};
      z = wmma_bf16(aq[0].v, b0, z);
      s[j] = wmma_bf16(aq[1].v, b1, z);
    }
    // ---- online softmax update (fp32) ----
    float cand[8], psum[8];
#pragma unroll
    for (int i = 0; i < 8; ++i) {
      s[0][i] *= scale;
      s[1][i] *= scale;
      cand[i] = fmaxf(s[0][i], s[1][i]);
    }
#pragma unroll
    for (int i = 0; i < 8; ++i) {
#pragma unroll
      for (int off = 1; off < 16; off <<= 1)
        cand[i] = fmaxf(cand[i], __shfl_xor(cand[i], off, 32));
    }
#pragma unroll
    for (int i = 0; i < 8; ++i) {
      float nm = fmaxf(mrow[i], cand[i]);
      float alpha = __expf(mrow[i] - nm);
      mrow[i] = nm;
      float p0 = __expf(s[0][i] - nm);
      float p1 = __expf(s[1][i] - nm);
      s[0][i] = p0;
      s[1][i] = p1;
      psum[i] = p0 + p1;
#pragma unroll
      for (int off = 1; off < 16; off <<= 1)
        psum[i] += __shfl_xor(psum[i], off, 32);
      lrow[i] = lrow[i] * alpha + psum[i];
      accO[0][i] *= alpha; accO[1][i] *= alpha;
      accO[2][i] *= alpha; accO[3][i] *= alpha;
    }
    // ---- C-layout -> A-layout transpose of P through LDS (bf16) ----
#pragma unroll
    for (int i = 0; i < 8; ++i) {
      int row = i + kh * 8;
      pbuf[row * 32 + nl] = (bf16)s[0][i];
      pbuf[row * 32 + 16 + nl] = (bf16)s[1][i];
    }
    AFrag pa;
    pa.h[0] = *(const v8bf*)(pbuf + nl * 32 + kh * 16);
    pa.h[1] = *(const v8bf*)(pbuf + nl * 32 + kh * 16 + 8);
    // ---- O += P * V, 4 DH sub-tiles ----
#pragma unroll
    for (int sub = 0; sub < 4; ++sub) {
      v16bf bv = *(const v16bf*)&Vs[cur][(sub * 16 + nl) * 32 + kh * 16];
      accO[sub] = wmma_bf16(pa.v, bv, accO[sub]);
    }
    __syncthreads();
  }

  const int b = bh >> 4, hh = bh & 15;
#pragma unroll
  for (int i = 0; i < 8; ++i) {
    float inv = 1.0f / lrow[i];
    int t = t0 + i + kh * 8;
    size_t rowoff = ((size_t)(b * CT + t)) * CD + hh * CDH;
#pragma unroll
    for (int sub = 0; sub < 4; ++sub)
      ctx[rowoff + sub * 16 + nl] = (bf16)(accO[sub][i] * inv);
  }
}

// ---------------------------------------------------------------------------
// Row LayerNorm over D=1024; one block per row; optional bf16 mirror.
// ---------------------------------------------------------------------------
template <bool WRITE_B>
__global__ __launch_bounds__(256) void ln_kernel(const float* __restrict__ y,
                                                 const float* __restrict__ g,
                                                 const float* __restrict__ be,
                                                 float* __restrict__ outf,
                                                 bf16* __restrict__ outb) {
  __shared__ float rs[8], rs2[8];
  const int row = blockIdx.x;
  const float* yr = y + (size_t)row * CD;
  float s = 0.0f, s2 = 0.0f, vals[4];
#pragma unroll
  for (int j = 0; j < 4; ++j) {
    float v = yr[threadIdx.x + j * 256];
    vals[j] = v;
    s += v;
    s2 += v * v;
  }
#pragma unroll
  for (int off = 16; off; off >>= 1) {
    s += __shfl_xor(s, off, 32);
    s2 += __shfl_xor(s2, off, 32);
  }
  const int wave = threadIdx.x >> 5, lane = threadIdx.x & 31;
  if (lane == 0) { rs[wave] = s; rs2[wave] = s2; }
  __syncthreads();
  if (wave == 0) {
    float a = (lane < 8) ? rs[lane] : 0.0f;
    float b2 = (lane < 8) ? rs2[lane] : 0.0f;
#pragma unroll
    for (int off = 4; off; off >>= 1) {
      a += __shfl_xor(a, off, 32);
      b2 += __shfl_xor(b2, off, 32);
    }
    if (lane == 0) { rs[0] = a; rs2[0] = b2; }
  }
  __syncthreads();
  float mean = rs[0] * (1.0f / CD);
  float var = rs2[0] * (1.0f / CD) - mean * mean;
  float rstd = rsqrtf(var + 1e-5f);
#pragma unroll
  for (int j = 0; j < 4; ++j) {
    int c = threadIdx.x + j * 256;
    float o = (vals[j] - mean) * rstd * g[c] + be[c];
    outf[(size_t)row * CD + c] = o;
    if (WRITE_B) outb[(size_t)row * CD + c] = (bf16)o;
  }
}

// ---------------------------------------------------------------------------
// launch
// ---------------------------------------------------------------------------
extern "C" void kernel_launch(void* const* d_in, const int* in_sizes, int n_in,
                              void* d_out, int out_size, void* d_ws,
                              size_t ws_size, hipStream_t stream) {
  (void)in_sizes; (void)n_in; (void)out_size; (void)ws_size;
  const float* x    = (const float*)d_in[0];
  const float* Wqkv = (const float*)d_in[1];
  const float* bqkv = (const float*)d_in[2];
  const float* Wo   = (const float*)d_in[3];
  const float* bo   = (const float*)d_in[4];
  const float* W1   = (const float*)d_in[5];
  const float* b1   = (const float*)d_in[6];
  const float* W2   = (const float*)d_in[7];
  const float* b2   = (const float*)d_in[8];
  const float* g1   = (const float*)d_in[9];
  const float* be1  = (const float*)d_in[10];
  const float* g2   = (const float*)d_in[11];
  const float* be2  = (const float*)d_in[12];

  // workspace layout (needs ~312 MiB; lifetimes allow heavy reuse)
  const size_t MiB = 1ull << 20;
  char* ws = (char*)d_ws;
  bf16*  xb    = (bf16*)(ws + 0);           // 32 MiB  (dead after QKV gemm)
  bf16*  qb    = (bf16*)(ws + 32 * MiB);    // 32 MiB
  bf16*  kb    = (bf16*)(ws + 64 * MiB);    // 32 MiB
  bf16*  vtb   = (bf16*)(ws + 96 * MiB);    // 32 MiB
  bf16*  ctxb  = (bf16*)(ws + 0);           // reuse xb slot
  bf16*  hb    = (bf16*)(ws + 0);           // 128 MiB, reuse xb+q+k+vt slots
  float* yb    = (float*)(ws + 128 * MiB);  // 64 MiB (pre-LN1 and pre-LN2)
  float* x1f   = (float*)(ws + 192 * MiB);  // 64 MiB
  bf16*  x1b   = (bf16*)(ws + 256 * MiB);   // 32 MiB
  bf16*  wqkvt = (bf16*)(ws + 288 * MiB);   // 6 MiB   [3D][D]
  bf16*  wot   = (bf16*)(ws + 294 * MiB);   // 2 MiB   [D][D]
  bf16*  w1t   = (bf16*)(ws + 296 * MiB);   // 8 MiB   [F][D]
  bf16*  w2t   = (bf16*)(ws + 304 * MiB);   // 8 MiB   [D][F]

  // 1) x -> bf16
  cvt_kernel<<<(CM * CD) / 1024, 256, 0, stream>>>(x, xb, (CM * CD) / 4);
  // 2) weight transposes (fp32 -> bf16, [K][N] -> [N][K])
  transpose_kernel<<<dim3(3 * CD / 32, CD / 32), 256, 0, stream>>>(Wqkv, wqkvt, CD, 3 * CD);
  transpose_kernel<<<dim3(CD / 32, CD / 32), 256, 0, stream>>>(Wo, wot, CD, CD);
  transpose_kernel<<<dim3(CF / 32, CD / 32), 256, 0, stream>>>(W1, w1t, CD, CF);
  transpose_kernel<<<dim3(CD / 32, CF / 32), 256, 0, stream>>>(W2, w2t, CF, CD);
  // 3) QKV projection, scatter into Q / K / V^T
  gemm_kernel<0><<<dim3(3 * CD / 128, CM / 64), 256, 0, stream>>>(
      xb, wqkvt, bqkv, qb, nullptr, kb, vtb, CM, 3 * CD, CD);
  // 4) flash attention -> ctx bf16 [B*T][D]
  attn_kernel<<<CB * CH * (CT / 128), 256, 0, stream>>>(qb, kb, vtb, ctxb);
  // 5) out-proj + residual(x) -> y fp32
  gemm_kernel<1><<<dim3(CD / 128, CM / 64), 256, 0, stream>>>(
      ctxb, wot, bo, yb, x, nullptr, nullptr, CM, CD, CD);
  // 6) LN1 -> x1 fp32 + bf16
  ln_kernel<true><<<CM, 256, 0, stream>>>(yb, g1, be1, x1f, x1b);
  // 7) FFN1: gelu(x1 @ W1 + b1) -> h bf16
  gemm_kernel<2><<<dim3(CF / 128, CM / 64), 256, 0, stream>>>(
      x1b, w1t, b1, hb, nullptr, nullptr, nullptr, CM, CF, CD);
  // 8) FFN2: h @ W2 + b2 + x1 -> y fp32
  gemm_kernel<1><<<dim3(CD / 128, CM / 64), 256, 0, stream>>>(
      hb, w2t, b2, yb, x1f, nullptr, nullptr, CM, CD, CF);
  // 9) LN2 -> out fp32
  ln_kernel<false><<<CM, 256, 0, stream>>>(yb, g2, be2, (float*)d_out, nullptr);
}
